// MultiHeadAttention_86122684220213
// MI455X (gfx1250) — compile-verified, matrix operand order checked by reading, not run on hardware
//
#include <hip/hip_runtime.h>

// ---- problem constants -------------------------------------------------
#define B_   2
#define C_   256
#define N_   3136      // 56*56
#define NH   8
#define D_   32
#define SCALE 0.17677669529663687f   // 1/sqrt(32), folded into Q

typedef __attribute__((ext_vector_type(16))) __bf16 v16bf;
typedef __attribute__((ext_vector_type(8)))  float  v8f;
typedef __attribute__((ext_vector_type(2)))  __bf16 v2bf;

union ABFrag { v16bf v; uint4 q[2]; };
union PairU  { v2bf v; unsigned int u; };

// Scheduling-directive helper (amdgcn sched_group_barrier):
//   0x008 = MFMA/WMMA group, 0x020 = VMEM read, 0x100 = DS read
#define SG(mask, size) __builtin_amdgcn_sched_group_barrier((mask), (size), 0)

// Load one WMMA A/B fragment following the CDNA5 16-bit operand layout:
// lanes 0-15 hold K {0..7,16..23}, lanes 16-31 hold K {8..15,24..31}.
// Caller passes p already offset by (lane>>4)*8 elements; we grab two
// 16-byte chunks 32 bytes apart.
static __device__ __forceinline__ v16bf load_frag(const __bf16* p) {
  ABFrag f;
  f.q[0] = *(const uint4*)p;
  f.q[1] = *(const uint4*)(p + 16);
  return f.v;
}

static __device__ __forceinline__ v8f wmma_bf16(v16bf a, v16bf b, v8f c) {
  return __builtin_amdgcn_wmma_f32_16x16x32_bf16(false, a, false, b,
                                                 (short)0, c, false, false);
}

// ---- K0: fp32 weights -> bf16 copies in workspace ----------------------
__global__ __launch_bounds__(256)
void wconv_kernel(const float* __restrict__ wqkv, const float* __restrict__ wproj,
                  __bf16* __restrict__ wqkvb, __bf16* __restrict__ wprojb) {
  int i = blockIdx.x * 256 + threadIdx.x;
  if (i < 768 * 256) wqkvb[i] = (__bf16)wqkv[i];
  int j = i - 768 * 256;
  if (j >= 0 && j < 256 * 256) wprojb[j] = (__bf16)wproj[j];
}

// ---- K1: QKV = X^T @ Wqkv^T, scatter to q/k/v [B,h,N,d] bf16 -----------
// grid (49, 2, 3): x = 64-row n tile, y = batch, z = 16 j-tile group.
__global__ __launch_bounds__(256)
void qkv_gemm_kernel(const float* __restrict__ x, const __bf16* __restrict__ wq,
                     __bf16* __restrict__ qb, __bf16* __restrict__ kb,
                     __bf16* __restrict__ vb) {
  __shared__ __align__(16) __bf16 sx[64 * 264];   // 64 rows, padded stride
  const int b = blockIdx.y, n0 = blockIdx.x * 64, zg = blockIdx.z;
  const int tid = threadIdx.x, lane = tid & 31, w = tid >> 5;
  const int half = lane >> 4, l16 = lane & 15;

  // transpose-stage x[b, :, n0..n0+63] into LDS as bf16 rows [n][c]
  for (int idx = tid; idx < 64 * 256; idx += 256) {
    int ni = idx & 63, c = idx >> 6;
    sx[ni * 264 + c] = (__bf16)x[((size_t)b * C_ + c) * N_ + (n0 + ni)];
  }
  __syncthreads();

  const int nsub = w & 3;
  const int arow = nsub * 16 + l16;
  v16bf A[8];
#pragma unroll
  for (int k = 0; k < 8; ++k)
    A[k] = load_frag(&sx[arow * 264 + k * 32 + half * 8]);

  for (int t = 0; t < 8; ++t) {
    int jt = zg * 16 + (w >> 2) + 2 * t;     // each wave: 8 of this group's 16 j-tiles
    int wrow = jt * 16 + l16;                // = output column j for this lane
    const __bf16* wp = wq + (size_t)wrow * 256 + half * 8;
    v16bf Bf[8];
#pragma unroll
    for (int k = 0; k < 8; ++k)
      Bf[k] = load_frag(wp + k * 32);
    v8f acc = {};
#pragma unroll
    for (int k = 0; k < 8; ++k)
      acc = wmma_bf16(A[k], Bf[k], acc);

    // Pin a depth-2 software pipeline: 4 loads (frags 0,1) up front, then
    // WMMA k overlapped with the 2 loads of frag k+2. Each WMMA then only
    // waits for its own fragment (loadcnt<=2), not a full drain.
    SG(0x020, 4);
#pragma unroll
    for (int k = 0; k < 6; ++k) { SG(0x008, 1); SG(0x020, 2); }
    SG(0x008, 1); SG(0x008, 1);

    int which = wrow >> 8;                   // 0=q 1=k 2=v
    int hh = (wrow >> 5) & 7, dd = wrow & 31;
    __bf16* dst = which == 0 ? qb : (which == 1 ? kb : vb);
    float mult = (which == 0) ? SCALE : 1.0f;  // fold softmax scale into Q
    size_t base = (size_t)(b * NH + hh) * N_ * D_ + dd;
#pragma unroll
    for (int r = 0; r < 8; ++r) {
      int n = n0 + nsub * 16 + half * 8 + r;
      dst[base + (size_t)n * D_] = (__bf16)(acc[r] * mult);
    }
  }
}

// ---- K2: flash attention per (b, h, 64-query tile) ---------------------
// 128 threads = 4 waves; wave w owns query rows [qt*64+w*16, +16).
// NOTE: the reference's edge boost is a per-query constant over keys, so
// softmax is invariant to it -> omitted entirely.
__global__ __launch_bounds__(128)
void attn_kernel(const __bf16* __restrict__ qb, const __bf16* __restrict__ kb,
                 const __bf16* __restrict__ vb, __bf16* __restrict__ ob) {
  __shared__ __align__(16) __bf16 sK[64 * 40];       // [key][d] padded
  __shared__ __align__(16) __bf16 sV[32 * 72];       // [d][key] (transposed)
  __shared__ __align__(16) float  sS[4 * 16 * 68];   // per-wave 16x64 scores
  __shared__ __align__(16) __bf16 sP[4 * 16 * 72];   // per-wave 16x64 probs
  const int qt = blockIdx.x, h = blockIdx.y, b = blockIdx.z;
  const int tid = threadIdx.x, lane = tid & 31, w = tid >> 5;
  const int half = lane >> 4, l16 = lane & 15;
  const size_t bh = (size_t)(b * NH + h) * N_ * D_;

  // Q fragment (16 rows x K=32) straight from global: rows are 64B-contig.
  v16bf qa = load_frag(qb + bh + (size_t)(qt * 64 + w * 16 + l16) * D_ + half * 8);

  v8f o0 = {}, o1 = {};
  float m_run = -1e30f, l_run = 0.f;
  float*  sSw = &sS[w * 16 * 68];
  __bf16* sPw = &sP[w * 16 * 72];

  for (int kbk = 0; kbk < 49; ++kbk) {
    const __bf16* kt = kb + bh + (size_t)kbk * 64 * D_;
    const __bf16* vt = vb + bh + (size_t)kbk * 64 * D_;
    // stage K tile [64 keys][32 d] (b128 loads)
    for (int i = tid; i < 256; i += 128) {
      int key = i >> 2, ch = i & 3;
      *(uint4*)&sK[key * 40 + ch * 8] = *(const uint4*)(kt + key * 32 + ch * 8);
    }
    // stage V transposed: [32 d][64 keys]
    for (int i = tid; i < 1024; i += 128) {
      int key = i >> 4, dp = i & 15;
      PairU p; p.u = *(const unsigned int*)(vt + key * 32 + dp * 2);
      sV[(2 * dp) * 72 + key]     = p.v[0];
      sV[(2 * dp + 1) * 72 + key] = p.v[1];
    }
    if (kbk + 1 < 49) {
      __builtin_prefetch(kt + 64 * D_ + (tid << 4), 0, 1);  // global_prefetch_b8
      __builtin_prefetch(vt + 64 * D_ + (tid << 4), 0, 1);
    }
    __syncthreads();

    // ---- S = Q @ K^T : preload all 4 B fragments, then 4 WMMAs
    v16bf kf[4];
#pragma unroll
    for (int ktile = 0; ktile < 4; ++ktile)
      kf[ktile] = load_frag(&sK[(ktile * 16 + l16) * 40 + half * 8]);
    v8f s[4];
#pragma unroll
    for (int ktile = 0; ktile < 4; ++ktile) {
      v8f z = {};
      s[ktile] = wmma_bf16(qa, kf[ktile], z);
    }
    // depth-2 pipeline on the LDS reads feeding the 4 S-tile WMMAs
    SG(0x100, 4);
    SG(0x008, 1); SG(0x100, 2);
    SG(0x008, 1); SG(0x100, 2);
    SG(0x008, 1); SG(0x008, 1);
#pragma unroll
    for (int ktile = 0; ktile < 4; ++ktile)
#pragma unroll
      for (int r = 0; r < 8; ++r)
        sSw[(r + 8 * half) * 68 + ktile * 16 + l16] = s[ktile][r];
    asm volatile("" ::: "memory");   // keep wave-internal DS ordering

    // ---- online softmax: lane -> (row l16, key-half half*32)
    const float* srow = sSw + l16 * 68 + half * 32;
    float4 tv[8];
#pragma unroll
    for (int j = 0; j < 8; ++j) tv[j] = *(const float4*)(srow + j * 4);
    float pm = -1e30f;
#pragma unroll
    for (int j = 0; j < 8; ++j)
      pm = fmaxf(pm, fmaxf(fmaxf(tv[j].x, tv[j].y), fmaxf(tv[j].z, tv[j].w)));
    float rmax = fmaxf(pm, __shfl_xor(pm, 16, 32));
    float mnew = fmaxf(m_run, rmax);
    float psum = 0.f;
    __bf16* pw = sPw + l16 * 72 + half * 32;
#pragma unroll
    for (int j = 0; j < 8; ++j) {
      float e0 = __expf(tv[j].x - mnew), e1 = __expf(tv[j].y - mnew);
      float e2 = __expf(tv[j].z - mnew), e3 = __expf(tv[j].w - mnew);
      psum += (e0 + e1) + (e2 + e3);
      PairU pA, pB;
      pA.v[0] = (__bf16)e0; pA.v[1] = (__bf16)e1;
      pB.v[0] = (__bf16)e2; pB.v[1] = (__bf16)e3;
      *(unsigned int*)(pw + j * 4)     = pA.u;
      *(unsigned int*)(pw + j * 4 + 2) = pB.u;
    }
    psum += __shfl_xor(psum, 16, 32);
    float alpha = __expf(m_run - mnew);
    l_run = l_run * alpha + psum;
    m_run = mnew;
    asm volatile("" ::: "memory");

    // ---- preload P (A) and V (B) fragments, then rescale O, then 4 WMMAs
    v16bf pa0 = load_frag(sPw + l16 * 72 + half * 8);
    v16bf pa1 = load_frag(sPw + l16 * 72 + 32 + half * 8);
    v16bf vf0 = load_frag(&sV[l16 * 72 + half * 8]);
    v16bf vf1 = load_frag(&sV[l16 * 72 + 32 + half * 8]);
    v16bf vf2 = load_frag(&sV[(16 + l16) * 72 + half * 8]);
    v16bf vf3 = load_frag(&sV[(16 + l16) * 72 + 32 + half * 8]);

    // rescale O accumulators (row alpha broadcast via ds_bpermute)
#pragma unroll
    for (int r = 0; r < 8; ++r) {
      float ar = __shfl(alpha, r + 8 * half, 32);
      o0[r] *= ar; o1[r] *= ar;
    }

    o0 = wmma_bf16(pa0, vf0, o0);
    o0 = wmma_bf16(pa1, vf1, o0);
    o1 = wmma_bf16(pa0, vf2, o1);
    o1 = wmma_bf16(pa1, vf3, o1);
    __syncthreads();
  }

  // normalize and write O as [B,N,C] bf16 (heads merged)
#pragma unroll
  for (int r = 0; r < 8; ++r) {
    int rr = r + 8 * half;
    float inv = 1.0f / __shfl(l_run, rr, 32);
    int n = qt * 64 + w * 16 + rr;
    size_t rowbase = ((size_t)b * N_ + n) * C_ + h * 32;
    ob[rowbase + l16]      = (__bf16)(o0[r] * inv);
    ob[rowbase + 16 + l16] = (__bf16)(o1[r] * inv);
  }
}

// ---- K3: out = O @ Wproj^T + b, stored transposed [B,C,N] fp32 ---------
// grid (49, 2, 2): x = 64-row n tile, y = batch, z = 8 j-tile group.
__global__ __launch_bounds__(256)
void proj_kernel(const __bf16* __restrict__ ob, const __bf16* __restrict__ wp,
                 const float* __restrict__ bias, float* __restrict__ out) {
  const int b = blockIdx.y, n0 = blockIdx.x * 64, zg = blockIdx.z;
  const int tid = threadIdx.x, lane = tid & 31, w = tid >> 5;
  const int half = lane >> 4, l16 = lane & 15;
  const int nsub = w & 3;
  const __bf16* oBase = ob + (size_t)b * N_ * C_;
  const int arow = n0 + nsub * 16 + l16;
  v16bf A[8];
#pragma unroll
  for (int k = 0; k < 8; ++k)   // O rows are contiguous -> no LDS staging
    A[k] = load_frag(oBase + (size_t)arow * C_ + k * 32 + half * 8);

  for (int t = 0; t < 4; ++t) {
    int jt = zg * 8 + (w >> 2) + 2 * t;
    int wrow = jt * 16 + l16;               // output channel c for this lane
    const __bf16* wpp = wp + (size_t)wrow * C_ + half * 8;
    v16bf Bf[8];
#pragma unroll
    for (int k = 0; k < 8; ++k)
      Bf[k] = load_frag(wpp + k * 32);
    v8f acc = {};
#pragma unroll
    for (int k = 0; k < 8; ++k)
      acc = wmma_bf16(A[k], Bf[k], acc);
    // same depth-2 pipeline as K1
    SG(0x020, 4);
#pragma unroll
    for (int k = 0; k < 6; ++k) { SG(0x008, 1); SG(0x020, 2); }
    SG(0x008, 1); SG(0x008, 1);

    float bv = bias[wrow];
    int nstart = n0 + nsub * 16 + half * 8;
    // C-layout rows are consecutive n -> contiguous fp32, pack as float4x2
    float* op = out + ((size_t)b * C_ + wrow) * N_ + nstart;
    float4 lo = make_float4(acc[0] + bv, acc[1] + bv, acc[2] + bv, acc[3] + bv);
    float4 hi = make_float4(acc[4] + bv, acc[5] + bv, acc[6] + bv, acc[7] + bv);
    *(float4*)op       = lo;
    *(float4*)(op + 4) = hi;
  }
}

// ---- launcher ----------------------------------------------------------
extern "C" void kernel_launch(void* const* d_in, const int* in_sizes, int n_in,
                              void* d_out, int out_size, void* d_ws, size_t ws_size,
                              hipStream_t stream) {
  const float* x     = (const float*)d_in[0];
  const float* Wqkv  = (const float*)d_in[1];
  const float* Wproj = (const float*)d_in[2];
  const float* bproj = (const float*)d_in[3];
  // d_in[4] (density_weight) intentionally unused: the edge boost is a
  // per-query additive constant over keys; softmax over keys is invariant.

  char* ws = (char*)d_ws;
  __bf16* qb   = (__bf16*)(ws + 0);          // B*h*N*d bf16 = 3,211,264 B each
  __bf16* kbuf = (__bf16*)(ws + 3211264);
  __bf16* vbuf = (__bf16*)(ws + 6422528);
  __bf16* obuf = (__bf16*)(ws + 9633792);    // [B,N,C] bf16
  __bf16* wqb  = (__bf16*)(ws + 12845056);   // Wqkv bf16
  __bf16* wpb  = (__bf16*)(ws + 13238272);   // Wproj bf16 (end ~13.37 MB)

  wconv_kernel<<<1024, 256, 0, stream>>>(Wqkv, Wproj, wqb, wpb);
  qkv_gemm_kernel<<<dim3(49, 2, 3), 256, 0, stream>>>(x, wqb, qb, kbuf, vbuf);
  attn_kernel<<<dim3(49, NH, 2), 128, 0, stream>>>(qb, kbuf, vbuf, obuf);
  proj_kernel<<<dim3(49, 2, 2), 256, 0, stream>>>(obuf, wpb, bproj, (float*)d_out);
}
